// GRU_12558484374152
// MI455X (gfx1250) — compile-verified, matrix operand order checked by reading, not run on hardware
//
#include <hip/hip_runtime.h>
#include <hip/hip_bf16.h>
#include <stdint.h>

// ---------------------------------------------------------------------------
// 2-layer GRU seq2seq (teacher forced), persistent-kernel implementation.
// bf16 WMMA (16x16x32) for all GEMMs, fp32 accumulation, fp32 hidden state.
// 64 persistent WGs; each WG owns 16 hidden columns; waves split K in half
// and combine partial accumulators through LDS. Two grid barriers per step.
// ---------------------------------------------------------------------------

typedef __bf16 bf16;
typedef __attribute__((ext_vector_type(16))) __bf16 v16bf;
typedef __attribute__((ext_vector_type(8)))  __bf16 v8bf;
typedef __attribute__((ext_vector_type(8)))  float  v8f;

constexpr int BATCH = 64;
constexpr int SLEN  = 512;
constexpr int TLEN  = 513;
constexpr int INP   = 256;
constexpr int OUTD  = 256;
constexpr int HID   = 1024;
constexpr int XW    = 512;   // concat(source, target) width
constexpr int NWG   = 64;    // persistent workgroups (co-resident)

// float -> bf16 with round-to-nearest-even
__device__ __forceinline__ bf16 tobf(float f) {
  unsigned u = __float_as_uint(f);
  u += 0x7FFFu + ((u >> 16) & 1u);
  unsigned short s = (unsigned short)(u >> 16);
  return __builtin_bit_cast(bf16, s);
}

__device__ __forceinline__ float fsigmoid(float x) {
  return 1.0f / (1.0f + __expf(-x));
}

// Load one lane's 16 bf16 fragment elements for a 16x16x32 WMMA k-step:
// two contiguous 16-byte groups 32 bytes apart (Ks {kg..kg+7},{kg+16..kg+23}).
__device__ __forceinline__ v16bf ldpair(const bf16* p) {
  v8bf lo = *(const v8bf*)(p);
  v8bf hi = *(const v8bf*)(p + 16);
  v16bf r;
#pragma unroll
  for (int i = 0; i < 8; ++i) { r[i] = lo[i]; r[i + 8] = hi[i]; }
  return r;
}

__device__ __forceinline__ v8f wmma_bf(v16bf a, v16bf b, v8f c) {
  return __builtin_amdgcn_wmma_f32_16x16x32_bf16(false, a, false, b,
                                                 (short)0, c, false, false);
}

// Grid-wide barrier over NWG workgroups via monotonically increasing counter.
__device__ __forceinline__ void gridbar(unsigned* bar, unsigned& epoch) {
  __syncthreads();
  __threadfence();
  ++epoch;
  if (threadIdx.x == 0) {
    atomicAdd(bar, 1u);
    const unsigned target = epoch * (unsigned)NWG;
    while (*(volatile unsigned*)bar < target) {
      __builtin_amdgcn_s_sleep(1);
    }
  }
  __syncthreads();
  __threadfence();
}

// ---------------------------------------------------------------------------
// Init: zero states / out[0] / barrier, convert weights to bf16, pack x=[s|t].
// ---------------------------------------------------------------------------
__global__ void gru_init(const float* __restrict__ src, const float* __restrict__ tgt,
                         const float* __restrict__ Wi0, const float* __restrict__ Wh0,
                         const float* __restrict__ Wi1, const float* __restrict__ Wh1,
                         const float* __restrict__ Wout,
                         bf16* __restrict__ xb,
                         bf16* __restrict__ cWi0, bf16* __restrict__ cWh0,
                         bf16* __restrict__ cWi1, bf16* __restrict__ cWh1,
                         bf16* __restrict__ cWout,
                         float* __restrict__ h0f, float* __restrict__ h1f,
                         bf16* __restrict__ h0b, bf16* __restrict__ h1b,
                         unsigned* __restrict__ bar, float* __restrict__ outp) {
  const size_t i0  = (size_t)blockIdx.x * blockDim.x + threadIdx.x;
  const size_t stp = (size_t)gridDim.x * blockDim.x;
  if (i0 == 0) *bar = 0u;
  for (size_t i = i0; i < (size_t)2 * BATCH * HID; i += stp) {
    h0f[i] = 0.f; h1f[i] = 0.f; h0b[i] = tobf(0.f); h1b[i] = tobf(0.f);
  }
  for (size_t i = i0; i < (size_t)BATCH * OUTD; i += stp) outp[i] = 0.f; // out[0]
  for (size_t i = i0; i < (size_t)3 * HID * XW; i += stp) cWi0[i] = tobf(Wi0[i]);
  for (size_t i = i0; i < (size_t)3 * HID * HID; i += stp) {
    cWh0[i] = tobf(Wh0[i]); cWi1[i] = tobf(Wi1[i]); cWh1[i] = tobf(Wh1[i]);
  }
  for (size_t i = i0; i < (size_t)OUTD * HID; i += stp) cWout[i] = tobf(Wout[i]);
  // xb[t][b][k] = k<INP ? source[b][t][k] : target[b][t][k-INP]  (bf16)
  for (size_t i = i0; i < (size_t)SLEN * BATCH * XW; i += stp) {
    const int    k  = (int)(i & (XW - 1));
    const size_t tb = i >> 9;                 // / XW
    const int    b  = (int)(tb & (BATCH - 1));
    const int    t  = (int)(tb >> 6);         // / BATCH
    const float  v  = (k < INP)
        ? src[((size_t)b * SLEN + t) * INP + k]
        : tgt[((size_t)b * TLEN + t) * OUTD + (k - INP)];
    xb[i] = tobf(v);
  }
}

// ---------------------------------------------------------------------------
// Persistent GRU sequence kernel: 64 WGs x 256 threads (8 wave32 per WG).
// wave w: mtile = w&3 (16 batch rows), khalf = w>>2 (K split).
// khalf=1 waves dump partial accumulators to LDS; khalf=0 waves combine and
// run the gate epilogue. Output head: 1 tile per WG, K split 8 ways + LDS
// reduction summed by wave 0.
// ---------------------------------------------------------------------------
__global__ __launch_bounds__(256) void gru_seq(
    const float* __restrict__ bi0, const float* __restrict__ bh0,
    const float* __restrict__ bi1, const float* __restrict__ bh1,
    const float* __restrict__ bout,
    const bf16* __restrict__ xb,
    const bf16* __restrict__ cWi0, const bf16* __restrict__ cWh0,
    const bf16* __restrict__ cWi1, const bf16* __restrict__ cWh1,
    const bf16* __restrict__ cWout,
    float* __restrict__ h0f, float* __restrict__ h1f,
    bf16* __restrict__ h0b, bf16* __restrict__ h1b,
    unsigned* __restrict__ bar, float* __restrict__ outp) {
  __shared__ float sred[3072];   // 12 KB: A/B phases use [mtile][acc][lane][8]

  const int lane  = threadIdx.x & 31;
  const int wid   = threadIdx.x >> 5;
  const int mtile = wid & 3;
  const int khalf = wid >> 2;
  const int l15   = lane & 15;
  const int kg    = (lane & 16) ? 8 : 0;
  const int mrow  = mtile * 16 + l15;                 // batch row for A loads
  const int colH  = blockIdx.x * 16 + l15;            // hidden column
  const int mbase = mtile * 16 + ((lane & 16) ? 8 : 0);

  unsigned epoch = 0;

  for (int t = 0; t < SLEN; ++t) {
    const int cur = t & 1, nxt = cur ^ 1;
    const unsigned h0cur = (unsigned)cur * BATCH * HID;
    const unsigned h0nxt = (unsigned)nxt * BATCH * HID;

    // ---------------- layer 0: h0' = GRU(x_t, h0) ----------------
    {
      v8f ar = {}, az = {}, ag = {}, ahn = {};
      if (khalf == 0) {
        // x part: K = [0,512) with Wi0 -> ar, az, ag(=i_n)
        {
          const bf16* arow = xb + (unsigned)(t * BATCH + mrow) * XW + kg;
          const bf16* wr = cWi0 + (unsigned)(0 * HID + colH) * XW + kg;
          const bf16* wz = cWi0 + (unsigned)(1 * HID + colH) * XW + kg;
          const bf16* wn = cWi0 + (unsigned)(2 * HID + colH) * XW + kg;
#pragma unroll 4
          for (int kk = 0; kk < XW; kk += 32) {
            v16bf a = ldpair(arow + kk);
            ar = wmma_bf(a, ldpair(wr + kk), ar);
            az = wmma_bf(a, ldpair(wz + kk), az);
            ag = wmma_bf(a, ldpair(wn + kk), ag);
          }
        }
        // h part: K = [0,256) with Wh0 -> ar, az, ahn
        {
          const bf16* arow = h0b + h0cur + (unsigned)mrow * HID + kg;
          const bf16* wr = cWh0 + (unsigned)(0 * HID + colH) * HID + kg;
          const bf16* wz = cWh0 + (unsigned)(1 * HID + colH) * HID + kg;
          const bf16* wn = cWh0 + (unsigned)(2 * HID + colH) * HID + kg;
#pragma unroll 4
          for (int kk = 0; kk < 256; kk += 32) {
            v16bf a = ldpair(arow + kk);
            ar  = wmma_bf(a, ldpair(wr + kk), ar);
            az  = wmma_bf(a, ldpair(wz + kk), az);
            ahn = wmma_bf(a, ldpair(wn + kk), ahn);
          }
        }
      } else {
        // h part: K = [256,1024) with Wh0 -> ar, az, ag(=h_n partial)
        const bf16* arow = h0b + h0cur + (unsigned)mrow * HID + 256 + kg;
        const bf16* wr = cWh0 + (unsigned)(0 * HID + colH) * HID + 256 + kg;
        const bf16* wz = cWh0 + (unsigned)(1 * HID + colH) * HID + 256 + kg;
        const bf16* wn = cWh0 + (unsigned)(2 * HID + colH) * HID + 256 + kg;
#pragma unroll 4
        for (int kk = 0; kk < 768; kk += 32) {
          v16bf a = ldpair(arow + kk);
          ar = wmma_bf(a, ldpair(wr + kk), ar);
          az = wmma_bf(a, ldpair(wz + kk), az);
          ag = wmma_bf(a, ldpair(wn + kk), ag);
        }
      }
      // combine K-halves through LDS
      if (khalf == 1) {
        float* d0 = &sred[((mtile * 3 + 0) * 32 + lane) * 8];
        float* d1 = &sred[((mtile * 3 + 1) * 32 + lane) * 8];
        float* d2 = &sred[((mtile * 3 + 2) * 32 + lane) * 8];
#pragma unroll
        for (int v = 0; v < 8; ++v) { d0[v] = ar[v]; d1[v] = az[v]; d2[v] = ag[v]; }
      }
      __syncthreads();
      if (khalf == 0) {
        const float* p0 = &sred[((mtile * 3 + 0) * 32 + lane) * 8];
        const float* p1 = &sred[((mtile * 3 + 1) * 32 + lane) * 8];
        const float* p2 = &sred[((mtile * 3 + 2) * 32 + lane) * 8];
        const float br  = bi0[colH] + bh0[colH];
        const float bz  = bi0[HID + colH] + bh0[HID + colH];
        const float bin = bi0[2 * HID + colH];
        const float bhn = bh0[2 * HID + colH];
        const float* hof = h0f + h0cur;
        float* hnf = h0f + h0nxt;
        bf16*  hnb = h0b + h0nxt;
#pragma unroll
        for (int v = 0; v < 8; ++v) {
          const unsigned m = mbase + v;
          const float hold = hof[m * HID + colH];
          const float r = fsigmoid(ar[v] + p0[v] + br);
          const float z = fsigmoid(az[v] + p1[v] + bz);
          const float n = tanhf(ag[v] + bin + r * (ahn[v] + p2[v] + bhn));
          const float hnew = (1.f - z) * n + z * hold;
          hnf[m * HID + colH] = hnew;
          hnb[m * HID + colH] = tobf(hnew);
        }
      }
    }
    gridbar(bar, epoch);

    // ---------------- layer 1: h1' = GRU(h0', h1) ----------------
    {
      v8f ar = {}, az = {}, ag = {};
      if (khalf == 0) {
        // input side: A = h0', W = Wi1 -> ar, az, ag(=i_n)
        const bf16* arow = h0b + h0nxt + (unsigned)mrow * HID + kg;
        const bf16* wr = cWi1 + (unsigned)(0 * HID + colH) * HID + kg;
        const bf16* wz = cWi1 + (unsigned)(1 * HID + colH) * HID + kg;
        const bf16* wn = cWi1 + (unsigned)(2 * HID + colH) * HID + kg;
#pragma unroll 4
        for (int kk = 0; kk < HID; kk += 32) {
          v16bf a = ldpair(arow + kk);
          ar = wmma_bf(a, ldpair(wr + kk), ar);
          az = wmma_bf(a, ldpair(wz + kk), az);
          ag = wmma_bf(a, ldpair(wn + kk), ag);
        }
      } else {
        // hidden side: A = h1, W = Wh1 -> ar, az, ag(=h_n)
        const bf16* arow = h1b + h0cur + (unsigned)mrow * HID + kg;
        const bf16* wr = cWh1 + (unsigned)(0 * HID + colH) * HID + kg;
        const bf16* wz = cWh1 + (unsigned)(1 * HID + colH) * HID + kg;
        const bf16* wn = cWh1 + (unsigned)(2 * HID + colH) * HID + kg;
#pragma unroll 4
        for (int kk = 0; kk < HID; kk += 32) {
          v16bf a = ldpair(arow + kk);
          ar = wmma_bf(a, ldpair(wr + kk), ar);
          az = wmma_bf(a, ldpair(wz + kk), az);
          ag = wmma_bf(a, ldpair(wn + kk), ag);
        }
      }
      if (khalf == 1) {
        float* d0 = &sred[((mtile * 3 + 0) * 32 + lane) * 8];
        float* d1 = &sred[((mtile * 3 + 1) * 32 + lane) * 8];
        float* d2 = &sred[((mtile * 3 + 2) * 32 + lane) * 8];
#pragma unroll
        for (int v = 0; v < 8; ++v) { d0[v] = ar[v]; d1[v] = az[v]; d2[v] = ag[v]; }
      }
      __syncthreads();
      if (khalf == 0) {
        const float* p0 = &sred[((mtile * 3 + 0) * 32 + lane) * 8];
        const float* p1 = &sred[((mtile * 3 + 1) * 32 + lane) * 8];
        const float* p2 = &sred[((mtile * 3 + 2) * 32 + lane) * 8];
        const float br  = bi1[colH] + bh1[colH];
        const float bz  = bi1[HID + colH] + bh1[HID + colH];
        const float bin = bi1[2 * HID + colH];
        const float bhn = bh1[2 * HID + colH];
        const float* hof = h1f + h0cur;
        float* hnf = h1f + h0nxt;
        bf16*  hnb = h1b + h0nxt;
#pragma unroll
        for (int v = 0; v < 8; ++v) {
          const unsigned m = mbase + v;
          const float hold = hof[m * HID + colH];
          const float r = fsigmoid(ar[v] + p0[v] + br);
          const float z = fsigmoid(az[v] + p1[v] + bz);
          const float n = tanhf(ag[v] + bin + r * (p2[v] + bhn));
          const float hnew = (1.f - z) * n + z * hold;
          hnf[m * HID + colH] = hnew;
          hnb[m * HID + colH] = tobf(hnew);
        }
      }
    }
    gridbar(bar, epoch);

    // ---------------- output head: out[t+1] = h1' @ Wout^T + bout ----------
    // One 16x16 tile per WG; 8 waves split K=1024 into 128-wide chunks,
    // summed by wave 0 through LDS. Overlaps into next step's layer 0.
    {
      const int mt = blockIdx.x & 3;
      const int nt = blockIdx.x >> 2;
      const int mr = mt * 16 + l15;
      const int co = nt * 16 + l15;
      v8f acc = {};
      const bf16* arow = h1b + h0nxt + (unsigned)mr * HID + (unsigned)wid * 128 + kg;
      const bf16* wrow = cWout + (unsigned)co * HID + (unsigned)wid * 128 + kg;
#pragma unroll
      for (int kk = 0; kk < 128; kk += 32) {
        acc = wmma_bf(ldpair(arow + kk), ldpair(wrow + kk), acc);
      }
      float* d = &sred[(wid * 32 + lane) * 8];
#pragma unroll
      for (int v = 0; v < 8; ++v) d[v] = acc[v];
      __syncthreads();
      if (wid == 0) {
#pragma unroll
        for (int w = 1; w < 8; ++w) {
          const float* p = &sred[(w * 32 + lane) * 8];
#pragma unroll
          for (int v = 0; v < 8; ++v) acc[v] += p[v];
        }
        const float bo = bout[co];
        const int mb = mt * 16 + ((lane & 16) ? 8 : 0);
#pragma unroll
        for (int v = 0; v < 8; ++v) {
          outp[((size_t)(t + 1) * BATCH + mb + v) * OUTD + co] = acc[v] + bo;
        }
      }
      __syncthreads();   // protect sred before next step's layer-0 writes
    }
  }
}

// ---------------------------------------------------------------------------
extern "C" void kernel_launch(void* const* d_in, const int* in_sizes, int n_in,
                              void* d_out, int out_size, void* d_ws, size_t ws_size,
                              hipStream_t stream) {
  (void)in_sizes; (void)n_in; (void)out_size; (void)ws_size;
  const float* src  = (const float*)d_in[0];
  const float* tgt  = (const float*)d_in[1];
  const float* Wi0  = (const float*)d_in[2];
  const float* Wh0  = (const float*)d_in[3];
  const float* bi0  = (const float*)d_in[4];
  const float* bh0  = (const float*)d_in[5];
  const float* Wi1  = (const float*)d_in[6];
  const float* Wh1  = (const float*)d_in[7];
  const float* bi1  = (const float*)d_in[8];
  const float* bh1  = (const float*)d_in[9];
  const float* Wout = (const float*)d_in[10];
  const float* bout = (const float*)d_in[11];
  float* outp = (float*)d_out;

  char* base = (char*)d_ws;
  size_t off = 0;
  auto carve = [&](size_t bytes) -> char* {
    char* q = base + off;
    off += (bytes + 255) & ~(size_t)255;
    return q;
  };
  unsigned* bar  = (unsigned*)carve(256);
  float* h0f = (float*)carve((size_t)2 * BATCH * HID * sizeof(float));
  float* h1f = (float*)carve((size_t)2 * BATCH * HID * sizeof(float));
  bf16*  h0b = (bf16*)carve((size_t)2 * BATCH * HID * sizeof(bf16));
  bf16*  h1b = (bf16*)carve((size_t)2 * BATCH * HID * sizeof(bf16));
  bf16*  xb  = (bf16*)carve((size_t)SLEN * BATCH * XW * sizeof(bf16));
  bf16*  cWi0 = (bf16*)carve((size_t)3 * HID * XW  * sizeof(bf16));
  bf16*  cWh0 = (bf16*)carve((size_t)3 * HID * HID * sizeof(bf16));
  bf16*  cWi1 = (bf16*)carve((size_t)3 * HID * HID * sizeof(bf16));
  bf16*  cWh1 = (bf16*)carve((size_t)3 * HID * HID * sizeof(bf16));
  bf16*  cWout = (bf16*)carve((size_t)OUTD * HID * sizeof(bf16));

  gru_init<<<512, 256, 0, stream>>>(src, tgt, Wi0, Wh0, Wi1, Wh1, Wout,
                                    xb, cWi0, cWh0, cWi1, cWh1, cWout,
                                    h0f, h1f, h0b, h1b, bar, outp);
  gru_seq<<<NWG, 256, 0, stream>>>(bi0, bh0, bi1, bh1, bout,
                                   xb, cWi0, cWh0, cWi1, cWh1, cWout,
                                   h0f, h1f, h0b, h1b, bar, outp);
}
